// Attention_23218593202739
// MI455X (gfx1250) — compile-verified, matrix operand order checked by reading, not run on hardware
//
#include <hip/hip_runtime.h>
#include <hip/hip_bf16.h>

// ---------------- CDNA5 WMMA types ----------------
typedef __attribute__((ext_vector_type(16))) __bf16        v16bf;
typedef __attribute__((ext_vector_type(8)))  float         v8f;
typedef __attribute__((ext_vector_type(8)))  unsigned int  v8u;

#define AQ   2048
#define AM   2048
#define DIM  1024
#define NH   16
#define HD   64      // per-head dim (both k and v)

// ---------------- helpers ----------------
__device__ __forceinline__ unsigned short f2bf(float f) {
    unsigned int u = __float_as_uint(f);
    u += 0x7FFFu + ((u >> 16) & 1u);   // round-to-nearest-even
    return (unsigned short)(u >> 16);
}

// Load one 16x32 bf16 A-fragment (or B^T-fragment) chunk pair from LDS.
// p points at row base + chunk offset; reads 16B at p and 16B at p+32B.
__device__ __forceinline__ v16bf frag_ld(const unsigned short* p) {
    uint4 lo = *(const uint4*)(p);
    uint4 hi = *(const uint4*)(p + 16);
    v8u t;
    t[0] = lo.x; t[1] = lo.y; t[2] = lo.z; t[3] = lo.w;
    t[4] = hi.x; t[5] = hi.y; t[6] = hi.z; t[7] = hi.w;
    return __builtin_bit_cast(v16bf, t);
}

__device__ __forceinline__ v8f wmma_bf16(v16bf a, v16bf b, v8f c) {
    return __builtin_amdgcn_wmma_f32_16x16x32_bf16(
        false, a, false, b, (short)0, c, false, false);
}

// =====================================================================
// Kernel 1: fused QKVG projections.
//   z=0: q  = (q_data @ Wq) * 0.125          -> bf16
//   z=1: k  =  m_data @ Wk                   -> bf16
//   z=2: v  =  m_data @ Wv                   -> bf16
//   z=3: g  = sigmoid(q_data @ Wg + bg)      -> f32
// Tile: 128x128, BK=32, 256 threads = 8 waves, each wave 64x32.
// =====================================================================
#define G_BM 128
#define G_BN 128
#define G_BK 32
#define G_LD 40   // LDS row stride (elements); 80B = multiple of 16B

__global__ __launch_bounds__(256) void proj_gemm(
    const float* __restrict__ q_data, const float* __restrict__ m_data,
    const float* __restrict__ Wq, const float* __restrict__ Wk,
    const float* __restrict__ Wv, const float* __restrict__ Wg,
    const float* __restrict__ bg,
    unsigned short* __restrict__ qb, unsigned short* __restrict__ kb,
    unsigned short* __restrict__ vb, float* __restrict__ gate)
{
    __shared__ __align__(16) unsigned short As[G_BM * G_LD];
    __shared__ __align__(16) unsigned short Bs[G_BN * G_LD];   // transposed [n][k]

    const int z = blockIdx.z;
    const float* A;
    const float* B;
    float scale = 1.0f;
    if      (z == 0) { A = q_data; B = Wq; scale = 0.125f; }
    else if (z == 1) { A = m_data; B = Wk; }
    else if (z == 2) { A = m_data; B = Wv; }
    else             { A = q_data; B = Wg; }

    const int N = DIM, K = DIM;
    const int bm = blockIdx.y * G_BM, bn = blockIdx.x * G_BN;
    const int tid = threadIdx.x, lane = tid & 31, wid = tid >> 5;
    const int wm = (wid & 1) * 64, wn = (wid >> 1) * 32;
    const int lh = lane >> 4, ln = lane & 15;
    const int c0 = lh * 8;

    v8f acc[4][2];
    for (int i = 0; i < 4; i++)
        for (int j = 0; j < 2; j++)
            for (int e = 0; e < 8; e++) acc[i][j][e] = 0.f;

    for (int kt = 0; kt < K; kt += G_BK) {
        __syncthreads();
        // A tile: 128x32 f32 -> bf16 LDS (row-major)
        #pragma unroll
        for (int it = 0; it < 4; it++) {
            int idx = tid + it * 256;          // 1024 float4 slots
            int r = idx >> 3, cv = idx & 7;
            float4 f = *(const float4*)(A + (size_t)(bm + r) * K + kt + cv * 4);
            ushort4 h;
            h.x = f2bf(f.x); h.y = f2bf(f.y); h.z = f2bf(f.z); h.w = f2bf(f.w);
            *(ushort4*)(&As[r * G_LD + cv * 4]) = h;
        }
        // B tile: 32x128 f32 -> bf16 LDS transposed [n][k]
        #pragma unroll
        for (int it = 0; it < 4; it++) {
            int idx = tid + it * 256;
            int kk = idx >> 5, nv = idx & 31;
            float4 f = *(const float4*)(B + (size_t)(kt + kk) * N + bn + nv * 4);
            Bs[(nv * 4 + 0) * G_LD + kk] = f2bf(f.x);
            Bs[(nv * 4 + 1) * G_LD + kk] = f2bf(f.y);
            Bs[(nv * 4 + 2) * G_LD + kk] = f2bf(f.z);
            Bs[(nv * 4 + 3) * G_LD + kk] = f2bf(f.w);
        }
        __syncthreads();

        v16bf af[4], bfv[2];
        #pragma unroll
        for (int mt = 0; mt < 4; mt++)
            af[mt] = frag_ld(&As[(wm + mt * 16 + ln) * G_LD + c0]);
        #pragma unroll
        for (int nt = 0; nt < 2; nt++)
            bfv[nt] = frag_ld(&Bs[(wn + nt * 16 + ln) * G_LD + c0]);
        #pragma unroll
        for (int mt = 0; mt < 4; mt++)
            #pragma unroll
            for (int nt = 0; nt < 2; nt++)
                acc[mt][nt] = wmma_bf16(af[mt], bfv[nt], acc[mt][nt]);
    }

    // epilogue (C layout: reg r -> row r + 8*lh, col = ln)
    #pragma unroll
    for (int mt = 0; mt < 4; mt++) {
        #pragma unroll
        for (int nt = 0; nt < 2; nt++) {
            int col = bn + wn + nt * 16 + ln;
            #pragma unroll
            for (int r = 0; r < 8; r++) {
                int row = bm + wm + mt * 16 + lh * 8 + r;
                float v = acc[mt][nt][r];
                if (z == 3) {
                    gate[(size_t)row * N + col] =
                        1.f / (1.f + __expf(-(v + bg[col])));
                } else {
                    unsigned short* dst = (z == 0) ? qb : ((z == 1) ? kb : vb);
                    dst[(size_t)row * N + col] = f2bf(v * scale);
                }
            }
        }
    }
}

// =====================================================================
// Kernel 2: flash attention per (head, 64-query block).
// 128 threads = 4 waves; wave w owns query rows w*16..w*16+15.
// Streams 64-key K/V blocks; pair_bias added to logits; online softmax.
// =====================================================================
#define SLD 72   // LDS row stride; 144B = multiple of 16B

__global__ __launch_bounds__(128) void attn_kernel(
    const unsigned short* __restrict__ qb, const unsigned short* __restrict__ kb,
    const unsigned short* __restrict__ vb, const float* __restrict__ gate,
    const float* __restrict__ pair_bias, unsigned short* __restrict__ attn)
{
    __shared__ __align__(16) unsigned short Qs[64 * SLD];
    __shared__ __align__(16) unsigned short Ks[64 * SLD];
    __shared__ __align__(16) unsigned short Vt[64 * SLD];       // [vd][key]
    __shared__ __align__(16) unsigned short Ps[4 * 16 * SLD];   // per-wave P

    const int h = blockIdx.y, qbase = blockIdx.x * 64;
    const int tid = threadIdx.x, lane = tid & 31, wid = tid >> 5;
    const int lh = lane >> 4, ln = lane & 15;
    const int c0 = lh * 8;

    // load Q tile (64 rows x 64 head-dims, bf16)
    #pragma unroll
    for (int it = 0; it < 8; it++) {
        int i = tid + it * 128;            // 1024 ushort4 slots
        int r = i >> 4, cv = i & 15;
        ushort4 d = *(const ushort4*)(qb + (size_t)(qbase + r) * DIM + h * HD + cv * 4);
        *(ushort4*)(&Qs[r * SLD + cv * 4]) = d;
    }

    v8f acc_o[4];
    for (int t = 0; t < 4; t++)
        for (int e = 0; e < 8; e++) acc_o[t][e] = 0.f;
    float m_i[8], l_i[8];
    for (int r = 0; r < 8; r++) { m_i[r] = -1e30f; l_i[r] = 0.f; }

    unsigned short* pw = &Ps[wid * 16 * SLD];

    for (int jb = 0; jb < AM / 64; jb++) {
        const int jbase = jb * 64;
        __syncthreads();
        // K block row-major, V block transposed into Vt[vd][key]
        #pragma unroll
        for (int it = 0; it < 8; it++) {
            int i = tid + it * 128;
            int r = i >> 4, cv = i & 15;
            ushort4 dk = *(const ushort4*)(kb + (size_t)(jbase + r) * DIM + h * HD + cv * 4);
            *(ushort4*)(&Ks[r * SLD + cv * 4]) = dk;
            ushort4 dv = *(const ushort4*)(vb + (size_t)(jbase + r) * DIM + h * HD + cv * 4);
            Vt[(cv * 4 + 0) * SLD + r] = dv.x;
            Vt[(cv * 4 + 1) * SLD + r] = dv.y;
            Vt[(cv * 4 + 2) * SLD + r] = dv.z;
            Vt[(cv * 4 + 3) * SLD + r] = dv.w;
        }
        __syncthreads();

        // S = Q K^T  (16x64 per wave; K rows ARE B^T rows -> no transpose)
        v16bf aq0 = frag_ld(&Qs[(wid * 16 + ln) * SLD + c0]);
        v16bf aq1 = frag_ld(&Qs[(wid * 16 + ln) * SLD + 32 + c0]);
        v8f s[4];
        #pragma unroll
        for (int nt = 0; nt < 4; nt++) {
            for (int e = 0; e < 8; e++) s[nt][e] = 0.f;
            v16bf b0 = frag_ld(&Ks[(nt * 16 + ln) * SLD + c0]);
            v16bf b1 = frag_ld(&Ks[(nt * 16 + ln) * SLD + 32 + c0]);
            s[nt] = wmma_bf16(aq0, b0, s[nt]);
            s[nt] = wmma_bf16(aq1, b1, s[nt]);
        }

        // + pair_bias[h, qrow, jcol]  (the dominant HBM stream, read once)
        #pragma unroll
        for (int nt = 0; nt < 4; nt++) {
            const float* bp = pair_bias
                + ((size_t)h * AQ + (qbase + wid * 16 + lh * 8)) * (size_t)AM
                + jbase + nt * 16 + ln;
            #pragma unroll
            for (int r = 0; r < 8; r++)
                s[nt][r] += bp[(size_t)r * AM];
        }

        // online softmax: row r lives in lanes {lh*16..lh*16+15}
        float alpha[8];
        #pragma unroll
        for (int r = 0; r < 8; r++) {
            float t = fmaxf(fmaxf(s[0][r], s[1][r]), fmaxf(s[2][r], s[3][r]));
            t = fmaxf(t, __shfl_xor(t, 1, 32));
            t = fmaxf(t, __shfl_xor(t, 2, 32));
            t = fmaxf(t, __shfl_xor(t, 4, 32));
            t = fmaxf(t, __shfl_xor(t, 8, 32));
            float mn = fmaxf(m_i[r], t);
            alpha[r] = __expf(m_i[r] - mn);
            m_i[r] = mn;
        }
        #pragma unroll
        for (int r = 0; r < 8; r++) {
            float rs = 0.f;
            #pragma unroll
            for (int nt = 0; nt < 4; nt++) {
                float p = __expf(s[nt][r] - m_i[r]);
                s[nt][r] = p;
                rs += p;
            }
            rs += __shfl_xor(rs, 1, 32);
            rs += __shfl_xor(rs, 2, 32);
            rs += __shfl_xor(rs, 4, 32);
            rs += __shfl_xor(rs, 8, 32);
            l_i[r] = l_i[r] * alpha[r] + rs;
        }

        // P (C layout) -> LDS -> A-fragment layout; rescale O; O += P V
        #pragma unroll
        for (int nt = 0; nt < 4; nt++)
            #pragma unroll
            for (int r = 0; r < 8; r++)
                pw[(lh * 8 + r) * SLD + nt * 16 + ln] = f2bf(s[nt][r]);
        #pragma unroll
        for (int vt = 0; vt < 4; vt++)
            #pragma unroll
            for (int r = 0; r < 8; r++)
                acc_o[vt][r] *= alpha[r];

        v16bf ap0 = frag_ld(&pw[ln * SLD + c0]);
        v16bf ap1 = frag_ld(&pw[ln * SLD + 32 + c0]);
        #pragma unroll
        for (int vt = 0; vt < 4; vt++) {
            v16bf b0 = frag_ld(&Vt[(vt * 16 + ln) * SLD + c0]);
            v16bf b1 = frag_ld(&Vt[(vt * 16 + ln) * SLD + 32 + c0]);
            acc_o[vt] = wmma_bf16(ap0, b0, acc_o[vt]);
            acc_o[vt] = wmma_bf16(ap1, b1, acc_o[vt]);
        }
    }

    // normalize, gate, store bf16
    #pragma unroll
    for (int r = 0; r < 8; r++) l_i[r] = 1.f / l_i[r];
    #pragma unroll
    for (int vt = 0; vt < 4; vt++) {
        int col = h * HD + vt * 16 + ln;
        #pragma unroll
        for (int r = 0; r < 8; r++) {
            int row = qbase + wid * 16 + lh * 8 + r;
            float o = acc_o[vt][r] * l_i[r];
            o *= gate[(size_t)row * DIM + col];
            attn[(size_t)row * DIM + col] = f2bf(o);
        }
    }
}

// =====================================================================
// Kernel 3: out = gatedO(bf16) @ Wo + bo  -> f32 d_out
// =====================================================================
__global__ __launch_bounds__(256) void out_gemm(
    const unsigned short* __restrict__ Abf, const float* __restrict__ Wo,
    const float* __restrict__ bo, float* __restrict__ out)
{
    __shared__ __align__(16) unsigned short As[G_BM * G_LD];
    __shared__ __align__(16) unsigned short Bs[G_BN * G_LD];

    const int N = DIM, K = DIM;
    const int bm = blockIdx.y * G_BM, bn = blockIdx.x * G_BN;
    const int tid = threadIdx.x, lane = tid & 31, wid = tid >> 5;
    const int wm = (wid & 1) * 64, wn = (wid >> 1) * 32;
    const int lh = lane >> 4, ln = lane & 15;
    const int c0 = lh * 8;

    v8f acc[4][2];
    for (int i = 0; i < 4; i++)
        for (int j = 0; j < 2; j++)
            for (int e = 0; e < 8; e++) acc[i][j][e] = 0.f;

    for (int kt = 0; kt < K; kt += G_BK) {
        __syncthreads();
        #pragma unroll
        for (int it = 0; it < 4; it++) {
            int idx = tid + it * 256;
            int r = idx >> 3, cv = idx & 7;
            ushort4 d = *(const ushort4*)(Abf + (size_t)(bm + r) * K + kt + cv * 4);
            *(ushort4*)(&As[r * G_LD + cv * 4]) = d;
        }
        #pragma unroll
        for (int it = 0; it < 4; it++) {
            int idx = tid + it * 256;
            int kk = idx >> 5, nv = idx & 31;
            float4 f = *(const float4*)(Wo + (size_t)(kt + kk) * N + bn + nv * 4);
            Bs[(nv * 4 + 0) * G_LD + kk] = f2bf(f.x);
            Bs[(nv * 4 + 1) * G_LD + kk] = f2bf(f.y);
            Bs[(nv * 4 + 2) * G_LD + kk] = f2bf(f.z);
            Bs[(nv * 4 + 3) * G_LD + kk] = f2bf(f.w);
        }
        __syncthreads();

        v16bf af[4], bfv[2];
        #pragma unroll
        for (int mt = 0; mt < 4; mt++)
            af[mt] = frag_ld(&As[(wm + mt * 16 + ln) * G_LD + c0]);
        #pragma unroll
        for (int nt = 0; nt < 2; nt++)
            bfv[nt] = frag_ld(&Bs[(wn + nt * 16 + ln) * G_LD + c0]);
        #pragma unroll
        for (int mt = 0; mt < 4; mt++)
            #pragma unroll
            for (int nt = 0; nt < 2; nt++)
                acc[mt][nt] = wmma_bf16(af[mt], bfv[nt], acc[mt][nt]);
    }

    #pragma unroll
    for (int mt = 0; mt < 4; mt++)
        #pragma unroll
        for (int nt = 0; nt < 2; nt++) {
            int col = bn + wn + nt * 16 + ln;
            float bias = bo[col];
            #pragma unroll
            for (int r = 0; r < 8; r++) {
                int row = bm + wm + mt * 16 + lh * 8 + r;
                out[(size_t)row * N + col] = acc[mt][nt][r] + bias;
            }
        }
}

// =====================================================================
// Host launcher
// =====================================================================
extern "C" void kernel_launch(void* const* d_in, const int* in_sizes, int n_in,
                              void* d_out, int out_size, void* d_ws, size_t ws_size,
                              hipStream_t stream) {
    const float* q_data    = (const float*)d_in[0];
    const float* m_data    = (const float*)d_in[1];
    // d_in[2] = bias (unused by module)
    const float* pair_bias = (const float*)d_in[3];
    const float* Wq        = (const float*)d_in[4];
    const float* Wk        = (const float*)d_in[5];
    const float* Wv        = (const float*)d_in[6];
    const float* Wg        = (const float*)d_in[7];
    const float* bg        = (const float*)d_in[8];
    const float* Wo        = (const float*)d_in[9];
    const float* bo        = (const float*)d_in[10];

    char* ws = (char*)d_ws;
    unsigned short* qb   = (unsigned short*)(ws);                     // 4 MiB bf16
    unsigned short* kb   = (unsigned short*)(ws + (4  << 20));        // 4 MiB bf16
    unsigned short* vb   = (unsigned short*)(ws + (8  << 20));        // 4 MiB bf16
    float*          gate = (float*)         (ws + (12 << 20));        // 8 MiB f32
    unsigned short* attn = (unsigned short*)(ws + (20 << 20));        // 4 MiB bf16

    // 1) fused Q/K/V/gate projections
    proj_gemm<<<dim3(DIM / G_BN, AQ / G_BM, 4), 256, 0, stream>>>(
        q_data, m_data, Wq, Wk, Wv, Wg, bg, qb, kb, vb, gate);

    // 2) flash attention + pair_bias + gating
    attn_kernel<<<dim3(AQ / 64, NH), 128, 0, stream>>>(
        qb, kb, vb, gate, pair_bias, attn);

    // 3) output projection
    out_gemm<<<dim3(DIM / G_BN, AQ / G_BM), 256, 0, stream>>>(
        attn, Wo, bo, (float*)d_out);
}
